// Classifier_82686710383325
// MI455X (gfx1250) — matrix operand order repaired and checked by reading, not compiled, not run on hardware
//
#include <hip/hip_runtime.h>

// ---------------------------------------------------------------------------
// SNN forward (snntorch-style Leaky) for MI455X / gfx1250, wave32 + WMMA f16.
//
// out layout (concatenated, f32):
//   spk_out [25,B,10] | spk1 [25,B,64] | spk2 [25,B,64] |
//   ss1 [25,B,64]     | ss2 [25,B,64]  | ss3 [25,B,10]
// ---------------------------------------------------------------------------

typedef __attribute__((ext_vector_type(16))) _Float16 v16h;
typedef __attribute__((ext_vector_type(8)))  float    v8f;

#define B_TOT   16384
#define NSTEPS  25
#define BETA    0.95f
#define THR     1.0f

union H16 { v16h v; float4 f[2]; };

static __device__ __forceinline__ v8f wmma_f16(v16h a, v16h b, v8f c) {
    // D(f32 16x16) = A(f16 16x32) x B(f16 32x16) + C
    return __builtin_amdgcn_wmma_f32_16x16x32_f16(
        /*neg_a=*/false, a, /*neg_b=*/false, b,
        /*c_mod=*/(short)0, c, /*reuse_a=*/false, /*reuse_b=*/false);
}

static __device__ __forceinline__ float fsigmoid(float t) {
    return 1.0f / (1.0f + __expf(-t));
}

// ---------------------------------------------------------------------------
// Kernel 1: x1 = x @ W1^T + b1   ([16384,784] x [64,784]^T -> [16384,64])
// One wave = 16 rows x 64 cols; K tiled in 25 chunks of 32 (tail zero-padded).
// ---------------------------------------------------------------------------
__global__ __launch_bounds__(256)
void snn_x1_gemm(const float* __restrict__ x, const float* __restrict__ W1,
                 const float* __restrict__ b1, float* __restrict__ x1) {
    const int lane = threadIdx.x & 31;
    const int wave = threadIdx.x >> 5;
    const int mrow = lane & 15;       // M index (lanes 0-15 / 16-31 both map M)
    const int khi  = lane >> 4;       // K half-select for A/B fragments
    const int rowbase = blockIdx.x * 128 + wave * 16;

    v8f acc[4] = {};                  // C/D tiles for n-groups 0..3

    const float* xrow = x + (size_t)(rowbase + mrow) * 784;

    for (int kg = 0; kg < 25; ++kg) {
        const int base1 = kg * 32 + khi * 16;  // A group1 K start (v0..3)
        const int base2 = base1 + 8;           // A group2 K start (v4..7)

        // prefetch next chunk of the streamed activations
        __builtin_prefetch(xrow + base1 + 64, 0, 1);

        float g1[8], g2[8];
        if (base1 < 784) {
            *(float4*)&g1[0] = *(const float4*)(xrow + base1);
            *(float4*)&g1[4] = *(const float4*)(xrow + base1 + 4);
        } else {
            #pragma unroll
            for (int i = 0; i < 8; ++i) g1[i] = 0.0f;
        }
        if (base2 < 784) {
            *(float4*)&g2[0] = *(const float4*)(xrow + base2);
            *(float4*)&g2[4] = *(const float4*)(xrow + base2 + 4);
        } else {
            #pragma unroll
            for (int i = 0; i < 8; ++i) g2[i] = 0.0f;
        }
        v16h a;
        #pragma unroll
        for (int i = 0; i < 8; ++i) { a[i] = (_Float16)g1[i]; a[8+i] = (_Float16)g2[i]; }

        const int k0 = kg * 32 + khi * 16;     // B fragment K start (16 halves)
        #pragma unroll
        for (int j = 0; j < 4; ++j) {
            const int n = j * 16 + mrow;
            const float* wrow = W1 + (size_t)n * 784;
            v16h bf;
            if (k0 < 784) {
                float w[16];
                *(float4*)&w[0]  = *(const float4*)(wrow + k0);
                *(float4*)&w[4]  = *(const float4*)(wrow + k0 + 4);
                *(float4*)&w[8]  = *(const float4*)(wrow + k0 + 8);
                *(float4*)&w[12] = *(const float4*)(wrow + k0 + 12);
                #pragma unroll
                for (int i = 0; i < 16; ++i) bf[i] = (_Float16)w[i];
            } else {
                #pragma unroll
                for (int i = 0; i < 16; ++i) bf[i] = (_Float16)0.0f;
            }
            acc[j] = wmma_f16(a, bf, acc[j]);
        }
    }

    // epilogue: + b1, store f32 (C/D layout: n = j*16+mrow, m = v + 8*khi)
    #pragma unroll
    for (int j = 0; j < 4; ++j) {
        const int n = j * 16 + mrow;
        const float bias = b1[n];
        #pragma unroll
        for (int v = 0; v < 8; ++v) {
            const int m = v + 8 * khi;
            x1[(size_t)(rowbase + m) * 64 + n] = acc[j][v] + bias;
        }
    }
}

// ---------------------------------------------------------------------------
// Kernel 2: 25-step LIF recurrence. One wave owns 16 batch rows; membrane
// potentials m1/m2/m3 live in VGPRs for the whole recurrence. Spikes are
// staged as f16 in wave-private LDS to re-layout C/D(M-major) -> A fragments.
// ---------------------------------------------------------------------------
__global__ __launch_bounds__(256)
void snn_steps(const float* __restrict__ x1,
               const float* __restrict__ W2, const float* __restrict__ b2,
               const float* __restrict__ W3, const float* __restrict__ b3,
               float* __restrict__ spk_out, float* __restrict__ spk1,
               float* __restrict__ spk2,    float* __restrict__ ss1,
               float* __restrict__ ss2,     float* __restrict__ ss3) {
    const int lane = threadIdx.x & 31;
    const int wave = threadIdx.x >> 5;
    const int mrow = lane & 15;
    const int khi  = lane >> 4;
    const int rowbase = blockIdx.x * 128 + wave * 16;

    __shared__ _Float16 sW2[64 * 64];      // W2 row-major [n][k], f16
    __shared__ _Float16 sW3[16 * 64];      // W3 padded to 16 rows, f16
    __shared__ _Float16 sS1[8][16 * 64];   // per-wave spike staging [m][k]
    __shared__ _Float16 sS2[8][16 * 64];

    for (int i = threadIdx.x; i < 64 * 64; i += 256) sW2[i] = (_Float16)W2[i];
    for (int i = threadIdx.x; i < 16 * 64; i += 256)
        sW3[i] = (i < 10 * 64) ? (_Float16)W3[i] : (_Float16)0.0f;
    __syncthreads();

    // load x1 tile + biases into registers (C/D layout)
    v8f   xr[4];
    float b2r[4];
    #pragma unroll
    for (int j = 0; j < 4; ++j) {
        const int n = j * 16 + mrow;
        b2r[j] = b2[n];
        #pragma unroll
        for (int v = 0; v < 8; ++v) {
            const int m = v + 8 * khi;
            xr[j][v] = x1[(size_t)(rowbase + m) * 64 + n];
        }
    }
    const float b3r   = (mrow < 10) ? b3[mrow] : 0.0f;
    const bool  nvalid = (mrow < 10);

    v8f m1[4] = {}, m2[4] = {}, m3 = {};
    _Float16* myS1 = sS1[wave];
    _Float16* myS2 = sS2[wave];

    for (int step = 0; step < NSTEPS; ++step) {
        const size_t so64 = (size_t)step * ((size_t)B_TOT * 64);
        const size_t so10 = (size_t)step * ((size_t)B_TOT * 10);

        // ---- LIF layer 1 (input x1, step-invariant) ----
        #pragma unroll
        for (int j = 0; j < 4; ++j) {
            const int n = j * 16 + mrow;
            #pragma unroll
            for (int v = 0; v < 8; ++v) {
                const int m = v + 8 * khi;
                float mem = BETA * m1[j][v] + xr[j][v];
                float spk = (mem > THR) ? 1.0f : 0.0f;
                float ssv = fsigmoid(mem - THR);
                m1[j][v] = mem - spk * THR;
                const size_t off = so64 + (size_t)(rowbase + m) * 64 + n;
                spk1[off] = spk;
                ss1[off]  = ssv;
                myS1[m * 64 + n] = (_Float16)spk;   // exact in f16
            }
        }

        // ---- h2 = s1 @ W2^T  (16x64x64 per wave, 8 WMMAs) ----
        v8f acc[4] = {};
        #pragma unroll
        for (int kg = 0; kg < 2; ++kg) {
            H16 af;
            const int ab = kg * 32 + khi * 16;
            af.f[0] = *(const float4*)(myS1 + mrow * 64 + ab);
            af.f[1] = *(const float4*)(myS1 + mrow * 64 + ab + 8);
            const int k0 = kg * 32 + khi * 16;
            #pragma unroll
            for (int j = 0; j < 4; ++j) {
                const int n = j * 16 + mrow;
                H16 bf;
                bf.f[0] = *(const float4*)(sW2 + n * 64 + k0);
                bf.f[1] = *(const float4*)(sW2 + n * 64 + k0 + 8);
                acc[j] = wmma_f16(af.v, bf.v, acc[j]);
            }
        }

        // ---- LIF layer 2 ----
        #pragma unroll
        for (int j = 0; j < 4; ++j) {
            const int n = j * 16 + mrow;
            #pragma unroll
            for (int v = 0; v < 8; ++v) {
                const int m = v + 8 * khi;
                float mem = BETA * m2[j][v] + (acc[j][v] + b2r[j]);
                float spk = (mem > THR) ? 1.0f : 0.0f;
                float ssv = fsigmoid(mem - THR);
                m2[j][v] = mem - spk * THR;
                const size_t off = so64 + (size_t)(rowbase + m) * 64 + n;
                spk2[off] = spk;
                ss2[off]  = ssv;
                myS2[m * 64 + n] = (_Float16)spk;
            }
        }

        // ---- h3 = s2 @ W3^T  (N padded 10->16 with zero rows, 2 WMMAs) ----
        v8f a3 = {};
        #pragma unroll
        for (int kg = 0; kg < 2; ++kg) {
            H16 af;
            const int ab = kg * 32 + khi * 16;
            af.f[0] = *(const float4*)(myS2 + mrow * 64 + ab);
            af.f[1] = *(const float4*)(myS2 + mrow * 64 + ab + 8);
            const int k0 = kg * 32 + khi * 16;
            H16 bf;
            bf.f[0] = *(const float4*)(sW3 + mrow * 64 + k0);
            bf.f[1] = *(const float4*)(sW3 + mrow * 64 + k0 + 8);
            a3 = wmma_f16(af.v, bf.v, a3);
        }

        // ---- LIF layer 3 (cols 0..9 valid) ----
        #pragma unroll
        for (int v = 0; v < 8; ++v) {
            const int m = v + 8 * khi;
            float mem = BETA * m3[v] + (a3[v] + b3r);
            float spk = (mem > THR) ? 1.0f : 0.0f;
            float ssv = fsigmoid(mem - THR);
            m3[v] = mem - spk * THR;
            if (nvalid) {
                const size_t off = so10 + (size_t)(rowbase + m) * 10 + mrow;
                spk_out[off] = spk;
                ss3[off]     = ssv;
            }
        }
    }
}

// ---------------------------------------------------------------------------
extern "C" void kernel_launch(void* const* d_in, const int* in_sizes, int n_in,
                              void* d_out, int out_size, void* d_ws, size_t ws_size,
                              hipStream_t stream) {
    const float* x  = (const float*)d_in[0];
    const float* W1 = (const float*)d_in[1];
    const float* b1 = (const float*)d_in[2];
    const float* W2 = (const float*)d_in[3];
    const float* b2 = (const float*)d_in[4];
    const float* W3 = (const float*)d_in[5];
    const float* b3 = (const float*)d_in[6];

    float* out = (float*)d_out;
    const size_t n10 = (size_t)NSTEPS * B_TOT * 10;
    const size_t n64 = (size_t)NSTEPS * B_TOT * 64;
    float* spk_out = out;
    float* spk1 = spk_out + n10;
    float* spk2 = spk1 + n64;
    float* ss1  = spk2 + n64;
    float* ss2  = ss1  + n64;
    float* ss3  = ss2  + n64;

    float* x1 = (float*)d_ws;  // [16384,64] f32 = 4 MB scratch

    snn_x1_gemm<<<dim3(128), dim3(256), 0, stream>>>(x, W1, b1, x1);
    snn_steps<<<dim3(128), dim3(256), 0, stream>>>(x1, W2, b2, W3, b3,
                                                   spk_out, spk1, spk2,
                                                   ss1, ss2, ss3);
}
